// cheb_conv_withSAt_73572789780805
// MI455X (gfx1250) — matrix operand order repaired and checked by reading, not compiled, not run on hardware
//
#include <hip/hip_runtime.h>
#include <hip/hip_bf16.h>
#include <stdint.h>

// Problem dims (fixed by the reference)
#define BB   16
#define NN   1024
#define FIN  64
#define TT   24
#define KC   3
#define FOUT 64
#define FT   1536          // FIN*TT == FOUT*TT

typedef __bf16 bf16;
typedef __attribute__((ext_vector_type(16))) __bf16    v16bf;
typedef __attribute__((ext_vector_type(8)))  float     v8f;
typedef __attribute__((ext_vector_type(8)))  uint32_t  v8u;

#if defined(__has_builtin)
#if __has_builtin(__builtin_amdgcn_tensor_load_to_lds) && __has_builtin(__builtin_amdgcn_s_wait_tensorcnt)
#define HAVE_TDM 1
#endif
#endif
#ifndef HAVE_TDM
#define HAVE_TDM 0
#endif

#if HAVE_TDM
typedef uint32_t u32x4 __attribute__((ext_vector_type(4)));
typedef int      i32x4 __attribute__((ext_vector_type(4)));
typedef int      i32x8 __attribute__((ext_vector_type(8)));

// LDS byte offset of a __shared__ object: hardware uses addr[31:0] of the
// flat shared address as the LDS address.
__device__ __forceinline__ uint32_t lds_off(const void* p) {
  return (uint32_t)(uintptr_t)p;
}

// TDM 2-D tile load: 128 rows x 32 bf16 (64B/row), row stride 1024 elements.
// D# per CDNA5 ISA ch.8 (groups 2/3 zero => <=2D tensor).
__device__ __forceinline__ void tdm_load_tile(uint32_t lds_addr, const bf16* gptr) {
  const uint64_t ga = (uint64_t)(uintptr_t)gptr;
  u32x4 g0;
  g0.x = 1u;                                   // count=1, user-mode, no gather
  g0.y = lds_addr;                             // lds_addr (bytes)
  g0.z = (uint32_t)ga;                         // global_addr[31:0]
  g0.w = (uint32_t)(ga >> 32) | (2u << 30);    // global_addr[56:32] | type=2
  i32x8 g1;
  g1[0] = (int)(1u << 16);        // workgroup_mask=0, data_size=1 (2 bytes)
  g1[1] = (int)(1024u << 16);     // tensor_dim0[15:0]=1024 (bits 63:48)
  g1[2] = (int)(1024u << 16);     // tensor_dim0 hi=0 | tensor_dim1[15:0]=1024
  g1[3] = (int)(32u << 16);       // tensor_dim1 hi=0 | tile_dim0=32 elems (64B)
  g1[4] = 128;                    // tile_dim1=128 rows, tile_dim2=0
  g1[5] = 1024;                   // tensor_dim0_stride lo (elements)
  g1[6] = 0;                      // stride hi=0, tensor_dim1_stride lo=0
  g1[7] = 0;
  i32x4 z4 = {0, 0, 0, 0};
#if __clang_major__ >= 23
  i32x8 z8 = {0, 0, 0, 0, 0, 0, 0, 0};
  __builtin_amdgcn_tensor_load_to_lds(g0, g1, z4, z4, z8, 0);
#else
  __builtin_amdgcn_tensor_load_to_lds(g0, g1, z4, z4, 0);
#endif
}
#endif  // HAVE_TDM

// ---------------------------------------------------------------------------
// K0: transpose x[b][n][f*T+t] (f32) -> Xbt[b][f][t][n] (bf16)
// ---------------------------------------------------------------------------
__global__ __launch_bounds__(256) void k_xpose(const float* __restrict__ x,
                                               bf16* __restrict__ Xbt) {
  __shared__ float tile[32][33];
  const int b  = blockIdx.z;
  const int n0 = blockIdx.y * 32;
  const int c0 = blockIdx.x * 32;     // ft index
  const int tx = threadIdx.x, ty = threadIdx.y;
#pragma unroll
  for (int j = 0; j < 4; ++j) {
    const int n = n0 + ty + j * 8;
    tile[ty + j * 8][tx] = x[((size_t)b * NN + n) * FT + c0 + tx];
  }
  __syncthreads();
#pragma unroll
  for (int j = 0; j < 4; ++j) {
    const int c = c0 + ty + j * 8;
    Xbt[((size_t)b * FT + c) * NN + n0 + tx] = (bf16)tile[tx][ty + j * 8];
  }
}

// ---------------------------------------------------------------------------
// K1: W[b,k][m][n] = cheb[k][n][m] * At[b][n][m]   (bf16, row-major in m)
// ---------------------------------------------------------------------------
__global__ __launch_bounds__(256) void k_buildW(const float* __restrict__ cheb,
                                                const float* __restrict__ At,
                                                bf16* __restrict__ W) {
  __shared__ float tile[32][33];
  const int bk = blockIdx.z;
  const int b  = bk / 3, k = bk - b * 3;
  const int m0 = blockIdx.x * 32, n0 = blockIdx.y * 32;
  const int tx = threadIdx.x, ty = threadIdx.y;
  const float* chb = cheb + (size_t)k * NN * NN;
  const float* at  = At   + (size_t)b * NN * NN;
#pragma unroll
  for (int j = 0; j < 4; ++j) {
    const int n = n0 + ty + j * 8;
    tile[ty + j * 8][tx] = chb[(size_t)n * NN + m0 + tx] * at[(size_t)n * NN + m0 + tx];
  }
  __syncthreads();
#pragma unroll
  for (int j = 0; j < 4; ++j) {
    const int m = m0 + ty + j * 8;
    W[((size_t)bk * NN + m) * NN + n0 + tx] = (bf16)tile[tx][ty + j * 8];
  }
}

// ---------------------------------------------------------------------------
// K2: Yt[b,k][o*T+t][n] = sum_f Theta[k][f][o] * Xbt[b][f][t][n]   (WMMA)
// ---------------------------------------------------------------------------
__global__ __launch_bounds__(256) void k_theta(const bf16* __restrict__ Xbt,
                                               const float* __restrict__ Theta,
                                               bf16* __restrict__ Yt) {
  const int b = blockIdx.z, t = blockIdx.y, nb = blockIdx.x;
  const int tid = threadIdx.x, lane = tid & 31, w = tid >> 5;
  const int o0 = (w >> 1) * 16;
  const int nbase = nb * 128 + (w & 1) * 64;

  v8f acc[KC][4];
#pragma unroll
  for (int k = 0; k < KC; ++k)
#pragma unroll
    for (int nt = 0; nt < 4; ++nt)
#pragma unroll
      for (int e = 0; e < 8; ++e) acc[k][nt][e] = 0.f;

#pragma unroll
  for (int fs = 0; fs < 2; ++fs) {
    v16bf bfr[4];
    const int fbase = fs * 32 + ((lane & 16) ? 16 : 0);
#pragma unroll
    for (int nt = 0; nt < 4; ++nt) {
      const int n = nbase + nt * 16 + (lane & 15);
      union { v16bf v; bf16 h[16]; } u;
#pragma unroll
      for (int e = 0; e < 16; ++e)
        u.h[e] = Xbt[(((size_t)b * FIN + fbase + e) * TT + t) * NN + n];
      bfr[nt] = u.v;
    }
    const int o = o0 + (lane & 15);
    const int abase = fs * 32 + ((lane & 16) ? 16 : 0);
#pragma unroll
    for (int k = 0; k < KC; ++k) {
      union { v16bf v; bf16 h[16]; } ua;
#pragma unroll
      for (int e = 0; e < 8; ++e) {
        ua.h[e]     = (bf16)Theta[((size_t)k * FIN + abase + e) * FOUT + o];
        ua.h[e + 8] = (bf16)Theta[((size_t)k * FIN + abase + 8 + e) * FOUT + o];
      }
#pragma unroll
      for (int nt = 0; nt < 4; ++nt)
        acc[k][nt] = __builtin_amdgcn_wmma_f32_16x16x32_bf16(
            false, ua.v, false, bfr[nt], (short)0, acc[k][nt], false, false);
    }
  }

  const int ncol = lane & 15;
  const int orow = (lane >> 4) * 8;
#pragma unroll
  for (int k = 0; k < KC; ++k)
#pragma unroll
    for (int nt = 0; nt < 4; ++nt) {
      const int n = nbase + nt * 16 + ncol;
#pragma unroll
      for (int v = 0; v < 8; ++v) {
        const int o = o0 + orow + v;
        Yt[((size_t)(b * KC + k) * FT + o * TT + t) * NN + n] = (bf16)acc[k][nt][v];
      }
    }
}

// ---------------------------------------------------------------------------
// K3: out[b][m][c] = relu( sum_k sum_n W[bk][m][n] * Yt[bk][c][n] )
// 128x128 C tile per block; 96 k-steps of 32. TDM (tensor_load_to_lds) feeds
// a 3-stage rotating LDS buffer: two transfers always in flight behind the
// WMMAs; TENSORcnt retires in order so s_wait_tensorcnt(2) pins stage it.
// ---------------------------------------------------------------------------
__global__ __launch_bounds__(256) void k_gemm(const bf16* __restrict__ W,
                                              const bf16* __restrict__ Yt,
                                              float* __restrict__ out) {
  __shared__ __align__(32) bf16 Asm[3][128 * 32];   // 8 KB per stage
  __shared__ __align__(32) bf16 Bsm[3][128 * 32];

  const int b  = blockIdx.z;
  const int m0 = blockIdx.y * 128;
  const int c0 = blockIdx.x * 128;
  const int tid = threadIdx.x, lane = tid & 31, w = tid >> 5;
  const int wm = w & 1, wc = w >> 1;
  const int lr = lane & 15, lhalf = lane >> 4;

  v8f acc[4][2];
#pragma unroll
  for (int i = 0; i < 4; ++i)
#pragma unroll
    for (int j = 0; j < 2; ++j)
#pragma unroll
      for (int e = 0; e < 8; ++e) acc[i][j][e] = 0.f;

  const size_t baseA = (size_t)(b * KC) * NN * NN;
  const size_t baseB = (size_t)(b * KC) * FT * NN;

#if HAVE_TDM
  // -------- Prologue: two stages in flight --------
  if (tid < 32) {
#pragma unroll
    for (int p = 0; p < 2; ++p) {
      const int kk = p >> 5;               // 0
      const int nn0 = (p & 31) << 5;
      const size_t aoff = baseA + (size_t)kk * NN * NN + nn0;
      const size_t boff = baseB + (size_t)kk * FT * NN + nn0;
      tdm_load_tile(lds_off(&Asm[p][0]), W  + aoff + (size_t)m0 * NN);
      tdm_load_tile(lds_off(&Bsm[p][0]), Yt + boff + (size_t)c0 * NN);
    }
  }
#else
  const int lrow = tid >> 1;
  const int lpart = (tid & 1) * 16;        // element offset within a 64B row
  {
    v8u a  = *(const v8u*)(W  + baseA + (size_t)(m0 + lrow) * NN + lpart);
    v8u bq = *(const v8u*)(Yt + baseB + (size_t)(c0 + lrow) * NN + lpart);
    *(v8u*)&Asm[0][lrow * 32 + lpart] = a;
    *(v8u*)&Bsm[0][lrow * 32 + lpart] = bq;
  }
  __syncthreads();
#endif

  int stage = 0;                           // = it % 3
  for (int it = 0; it < 96; ++it) {
    const int itn = it + 1;

#if HAVE_TDM
    // Pin stage `it`: oldest outstanding pair must have landed.  While
    // iterations remain, allow T(it+1)'s two loads to stay in flight.
    if (tid < 32) {
      if (itn < 96) __builtin_amdgcn_s_wait_tensorcnt(2);
      else          __builtin_amdgcn_s_wait_tensorcnt(0);
    }
    __syncthreads();   // publish stage `it`; also fences reuse of stage it-1

    // Kick off T(it+2) into the stage last read at iteration it-1.
    if (it + 2 < 96 && tid < 32) {
      const int i2 = it + 2;
      const int kk = i2 >> 5;
      const int nn0 = (i2 & 31) << 5;
      const size_t aoff = baseA + (size_t)kk * NN * NN + nn0;
      const size_t boff = baseB + (size_t)kk * FT * NN + nn0;
      const int s2 = (stage + 2 >= 3) ? stage - 1 : stage + 2;
      tdm_load_tile(lds_off(&Asm[s2][0]), W  + aoff + (size_t)m0 * NN);
      tdm_load_tile(lds_off(&Bsm[s2][0]), Yt + boff + (size_t)c0 * NN);
    }
#else
    v8u na = {}, nb = {};
    if (itn < 96) {
      const int kk = itn >> 5;
      const int nn0 = (itn & 31) << 5;
      const size_t aoff = baseA + (size_t)kk * NN * NN + nn0;
      const size_t boff = baseB + (size_t)kk * FT * NN + nn0;
      na = *(const v8u*)(W  + aoff + (size_t)(m0 + lrow) * NN + lpart);
      nb = *(const v8u*)(Yt + boff + (size_t)(c0 + lrow) * NN + lpart);
    }
#endif

    // -------- Compute on stage `it` --------
    // A fragment: row m, 16 contiguous k (lanes 0-15: k0-15, lanes 16-31: k16-31).
    // B fragment: row c, 16 contiguous n (low lanes n0-15, high lanes n16-31).
    v16bf afr[4], bfr[2];
#pragma unroll
    for (int i = 0; i < 4; ++i) {
      const bf16* arow = &Asm[stage][((wm * 4 + i) * 16 + lr) * 32 + (lhalf ? 16 : 0)];
      union { uint4 q[2]; v16bf v; } u;
      u.q[0] = *(const uint4*)(arow);
      u.q[1] = *(const uint4*)(arow + 8);
      afr[i] = u.v;
    }
#pragma unroll
    for (int j = 0; j < 2; ++j) {
      const bf16* brow = &Bsm[stage][((wc * 2 + j) * 16 + lr) * 32 + (lhalf ? 16 : 0)];
      bfr[j] = __builtin_bit_cast(v16bf, *(const v8u*)brow);
    }
#pragma unroll
    for (int i = 0; i < 4; ++i)
#pragma unroll
      for (int j = 0; j < 2; ++j)
        acc[i][j] = __builtin_amdgcn_wmma_f32_16x16x32_bf16(
            false, afr[i], false, bfr[j], (short)0, acc[i][j], false, false);

#if HAVE_TDM
    stage = (stage + 1 >= 3) ? 0 : stage + 1;
#else
    __syncthreads();
    if (itn < 96) {
      stage ^= 1;
      *(v8u*)&Asm[stage][lrow * 32 + lpart] = na;
      *(v8u*)&Bsm[stage][lrow * 32 + lpart] = nb;
      __syncthreads();
    }
#endif
  }

  // -------- Epilogue: fused ReLU, coalesced f32 stores --------
  float* outb = out + (size_t)b * NN * FT;
  const int ccol = lane & 15;
  const int moff = (lane >> 4) * 8;
#pragma unroll
  for (int i = 0; i < 4; ++i) {
    const int mbase = m0 + (wm * 4 + i) * 16;
#pragma unroll
    for (int j = 0; j < 2; ++j) {
      const int cbase = c0 + (wc * 2 + j) * 16;
#pragma unroll
      for (int v = 0; v < 8; ++v) {
        const float r = acc[i][j][v];
        outb[(size_t)(mbase + moff + v) * FT + cbase + ccol] = r > 0.f ? r : 0.f;
      }
    }
  }
}

// ---------------------------------------------------------------------------
// Workspace layout (240 MB total):
//   [0, 151.0 MB)   Yt  bf16  (3*16 x 1536 x 1024)
//   [151.0, 251.7)  W   bf16  (48 x 1024 x 1024); Xbt (50.3 MB) aliases W's
//                   region and is consumed by k_theta before k_buildW runs.
// ---------------------------------------------------------------------------
extern "C" void kernel_launch(void* const* d_in, const int* in_sizes, int n_in,
                              void* d_out, int out_size, void* d_ws, size_t ws_size,
                              hipStream_t stream) {
  const float* x     = (const float*)d_in[0];
  const float* At    = (const float*)d_in[1];
  const float* cheb  = (const float*)d_in[2];
  const float* Theta = (const float*)d_in[3];
  float* out = (float*)d_out;

  const size_t YT_ELEMS = (size_t)BB * KC * FT * NN;   // 75,497,472
  bf16* Yt  = (bf16*)d_ws;
  bf16* Wt  = (bf16*)((char*)d_ws + YT_ELEMS * sizeof(bf16));
  bf16* Xbt = Wt;  // alias: dead before k_buildW overwrites it

  k_xpose <<<dim3(FT / 32, NN / 32, BB),      dim3(32, 8), 0, stream>>>(x, Xbt);
  k_theta <<<dim3(NN / 128, TT, BB),          256,         0, stream>>>(Xbt, Theta, Yt);
  k_buildW<<<dim3(NN / 32, NN / 32, BB * KC), dim3(32, 8), 0, stream>>>(cheb, At, Wt);
  k_gemm  <<<dim3(FT / 128, NN / 128, BB),    256,         0, stream>>>(Wt, Yt, out);
}
